// GATS_72645076844636
// MI455X (gfx1250) — matrix operand order repaired and checked by reading, not compiled, not run on hardware
//
#include <hip/hip_runtime.h>
#include <hip/hip_bf16.h>
#include <hip/hip_fp16.h>

typedef __attribute__((ext_vector_type(16))) _Float16 v16h;
typedef __attribute__((ext_vector_type(8)))  _Float16 v8h;
typedef __attribute__((ext_vector_type(8)))  float    v8f;

#define NEG_SLOPE 0.2f

__device__ __forceinline__ float softplus_f(float v) {
    // numerically stable softplus
    return fmaxf(v, 0.0f) + log1pf(__expf(-fabsf(v)));
}

__device__ __forceinline__ unsigned float_to_ordered_key(float f) {
    unsigned k = __float_as_uint(f);
    return (k & 0x80000000u) ? ~k : (k | 0x80000000u);
}
__device__ __forceinline__ float ordered_key_to_float(unsigned k) {
    return (k & 0x80000000u) ? __uint_as_float(k ^ 0x80000000u) : __uint_as_float(~k);
}

// ---------------------------------------------------------------- init
__global__ void zero_ws(unsigned* __restrict__ p, long n) {
    long i = (long)blockIdx.x * blockDim.x + threadIdx.x;
    if (i < n) p[i] = 0u;
}

// ---------------------------------------------------------------- per-node prep
// min/max normalize, 64-elem bitonic sort (static indices -> VGPRs),
// conf = max softmax = 1/sum(exp(x - xmax)), a_cluster + reciprocal,
// sorted row stored as f16 (row-major, read later in 8-half chunks).
__global__ void node_prep(const float* __restrict__ x,
                          const int*   __restrict__ dist,
                          const float* __restrict__ train_a,
                          const float* __restrict__ dist1_a,
                          _Float16*    __restrict__ xs,
                          float*       __restrict__ a_cl,
                          float*       __restrict__ rcp_a,
                          float*       __restrict__ conf,
                          int N)
{
    int n = blockIdx.x * blockDim.x + threadIdx.x;
    if (n >= N) return;

    const float4* xr = (const float4*)(x + (size_t)n * 64);
    float v[64];
#pragma unroll
    for (int i = 0; i < 16; ++i) {
        float4 t = xr[i];
        v[4*i+0] = t.x; v[4*i+1] = t.y; v[4*i+2] = t.z; v[4*i+3] = t.w;
    }
    float mn = v[0], mx = v[0];
#pragma unroll
    for (int i = 1; i < 64; ++i) { mn = fminf(mn, v[i]); mx = fmaxf(mx, v[i]); }
    float s = 0.f;
#pragma unroll
    for (int i = 0; i < 64; ++i) s += __expf(v[i] - mx);
    conf[n] = 1.0f / s;                       // exp(xmax - logsumexp)

    float inv = 1.0f / (mx - mn);
#pragma unroll
    for (int i = 0; i < 64; ++i) v[i] = (v[i] - mn) * inv;

    // bitonic sort, ascending; all indices constant after unroll
#pragma unroll
    for (int k = 2; k <= 64; k <<= 1) {
#pragma unroll
        for (int j = k >> 1; j > 0; j >>= 1) {
#pragma unroll
            for (int i = 0; i < 64; ++i) {
                int l = i ^ j;
                if (l > i) {
                    float a = v[i], b = v[l];
                    float lo = fminf(a, b), hi = fmaxf(a, b);
                    bool up = ((i & k) == 0);
                    v[i] = up ? lo : hi;
                    v[l] = up ? hi : lo;
                }
            }
        }
    }

    int d = dist[n];
    float a = (d == 0) ? train_a[0] : ((d == 1) ? dist1_a[0] : 1.0f);
    a_cl[n]  = a;
    rcp_a[n] = 1.0f / a;

    v8h* out = (v8h*)(xs + (size_t)n * 64);
#pragma unroll
    for (int c = 0; c < 8; ++c) {
        v8h p;
#pragma unroll
        for (int j = 0; j < 8; ++j) p[j] = (_Float16)v[8*c + j];
        out[c] = p;
    }
}

// ---------------------------------------------------------------- pad W_temp -> f16 [16][64]
__global__ void prep_w(const float* __restrict__ W, _Float16* __restrict__ Wpad) {
    int i = blockIdx.x * blockDim.x + threadIdx.x;
    if (i >= 16 * 64) return;
    int h = i >> 6, k = i & 63;
    Wpad[i] = (_Float16)((h < 8) ? W[h * 64 + k] : 0.0f);
}

// ---------------------------------------------------------------- WMMA: temp = xs @ Wpad^T, scaled
// One wave per 16-node tile. A = 16x64 f16 sorted rows, B = 64x16 f16 (heads padded to 16).
// Two v_wmma_f32_16x16x32_f16 accumulate K=0..31 and K=32..63.
__global__ void temp_wmma(const _Float16* __restrict__ xs,
                          const _Float16* __restrict__ Wpad,
                          const float*    __restrict__ a_cl,
                          float*          __restrict__ temp_scaled,
                          int N)
{
    int lane = threadIdx.x & 31;
    int tile = blockIdx.x * (blockDim.x >> 5) + (threadIdx.x >> 5);
    if (tile * 16 >= N) return;

    int m    = lane & 15;   // A: row M ; B: column N
    int half = lane >> 4;   // selects K subset per ISA layout

    // A fragment (16x32 f16): lane holds K in {half*8..+7} and {16+half*8..+7}
    const v8h* ax = (const v8h*)(xs + (size_t)(tile * 16 + m) * 64);
    v8h a_lo0 = ax[half],         a_hi0 = ax[2 + half];       // K0 = 0
    v8h a_lo1 = ax[4 + half],     a_hi1 = ax[6 + half];       // K0 = 32
    // B fragment (32x16 f16): lane holds K in {half*8..+7} and {16+half*8..+7} of column m
    const v8h* bx = (const v8h*)(Wpad + (size_t)m * 64);
    v8h b_lo0 = bx[half],         b_hi0 = bx[2 + half];       // K0 = 0
    v8h b_lo1 = bx[4 + half],     b_hi1 = bx[6 + half];       // K0 = 32

    v16h A0, A1, B0, B1;
#pragma unroll
    for (int i = 0; i < 8; ++i) {
        A0[i] = a_lo0[i]; A0[i + 8] = a_hi0[i];
        A1[i] = a_lo1[i]; A1[i + 8] = a_hi1[i];
        B0[i] = b_lo0[i]; B0[i + 8] = b_hi0[i];
        B1[i] = b_lo1[i]; B1[i + 8] = b_hi1[i];
    }

    v8f c = {};
    c = __builtin_amdgcn_wmma_f32_16x16x32_f16(false, A0, false, B0, (short)0, c, false, false);
    c = __builtin_amdgcn_wmma_f32_16x16x32_f16(false, A1, false, B1, (short)0, c, false, false);

    // D layout: lane -> column (lane&15); VGPR r -> row r + 8*(lane>>4)
    int col = lane & 15;
    int rowoff = half * 8;
    if (col < 8) {
#pragma unroll
        for (int r = 0; r < 8; ++r) {
            int node = tile * 16 + rowoff + r;
            if (node < N) {
                temp_scaled[(size_t)node * 8 + col] = c[r] * a_cl[node];
            }
        }
    }
}

// ---------------------------------------------------------------- edge pass 1: e, segment max, deg
__global__ void edge_pass1(const float* __restrict__ x,
                           const int*   __restrict__ src,
                           const int*   __restrict__ dst,
                           const float* __restrict__ rcp_a,
                           float*       __restrict__ e_out,
                           unsigned*    __restrict__ emax_key,
                           unsigned*    __restrict__ deg,
                           int E)
{
    int i = blockIdx.x * blockDim.x + threadIdx.x;
    if (i >= E) return;
    int s = src[i], d = dst[i];
    const float4* xa = (const float4*)(x + (size_t)s * 64);
    const float4* xb = (const float4*)(x + (size_t)d * 64);
    float acc = 0.f;
#pragma unroll
    for (int k = 0; k < 16; ++k) {
        float4 a = xa[k], b = xb[k];
        acc = fmaf(a.x, b.x, acc); acc = fmaf(a.y, b.y, acc);
        acc = fmaf(a.z, b.z, acc); acc = fmaf(a.w, b.w, acc);
    }
    acc *= rcp_a[s] * rcp_a[d];                 // (x_s/a_s)·(x_d/a_d)
    float e = (acc > 0.f) ? acc : NEG_SLOPE * acc;
    e_out[i] = e;
    atomicMax(&emax_key[d], float_to_ordered_key(e));
    atomicAdd(&deg[s], 1u);                     // degree counted over src (per reference)
}

// ---------------------------------------------------------------- edge pass 2: exp + segment sum
__global__ void edge_pass2(const int*      __restrict__ dst,
                           const unsigned* __restrict__ emax_key,
                           float*          __restrict__ e_buf,
                           float*          __restrict__ denom,
                           int E)
{
    int i = blockIdx.x * blockDim.x + threadIdx.x;
    if (i >= E) return;
    int d = dst[i];
    float m = ordered_key_to_float(emax_key[d]);
    float ex = __expf(e_buf[i] - m);
    e_buf[i] = ex;                              // reuse buffer
    unsafeAtomicAdd(&denom[d], ex);
}

// ---------------------------------------------------------------- edge pass 3: weighted scatter-add
__global__ void edge_pass3(const int*   __restrict__ src,
                           const int*   __restrict__ dst,
                           const float* __restrict__ e_buf,
                           const float* __restrict__ denom,
                           const float* __restrict__ temp_scaled,
                           const float* __restrict__ conf,
                           float*       __restrict__ sim,
                           float*       __restrict__ dconf,
                           int E)
{
    int i = blockIdx.x * blockDim.x + threadIdx.x;
    if (i >= E) return;
    int s = src[i], d = dst[i];
    float alpha = e_buf[i] / denom[d];
    const float4* ts = (const float4*)(temp_scaled + (size_t)s * 8);
    float4 t0 = ts[0], t1 = ts[1];
    float* so = sim + (size_t)d * 8;
    unsafeAtomicAdd(&so[0], t0.x * alpha);
    unsafeAtomicAdd(&so[1], t0.y * alpha);
    unsafeAtomicAdd(&so[2], t0.z * alpha);
    unsafeAtomicAdd(&so[3], t0.w * alpha);
    unsafeAtomicAdd(&so[4], t1.x * alpha);
    unsafeAtomicAdd(&so[5], t1.y * alpha);
    unsafeAtomicAdd(&so[6], t1.z * alpha);
    unsafeAtomicAdd(&so[7], t1.w * alpha);
    unsafeAtomicAdd(&dconf[d], conf[d] - conf[s]);
}

// ---------------------------------------------------------------- finalize: temperature, output
__global__ void finalize_k(const float*    __restrict__ x,
                           const float*    __restrict__ sim,
                           const float*    __restrict__ dconf,
                           const unsigned* __restrict__ deg,
                           const float*    __restrict__ conf_coef,
                           const float*    __restrict__ bias,
                           float*          __restrict__ out,
                           int N)
{
    int n = blockIdx.x * blockDim.x + threadIdx.x;
    if (n >= N) return;
    unsigned dg = deg[n];
    float dinv = dg ? 1.0f / (float)dg : 0.0f;
    float add = conf_coef[0] * dconf[n] * dinv;
    const float* sp = sim + (size_t)n * 8;
    float t = 0.f;
#pragma unroll
    for (int h = 0; h < 8; ++h) t += softplus_f(sp[h] + add);
    t = t * 0.125f + bias[0];
    float rt = 1.0f / t;
    const float4* xi = (const float4*)(x + (size_t)n * 64);
    float4* xo = (float4*)(out + (size_t)n * 64);
#pragma unroll
    for (int i = 0; i < 16; ++i) {
        float4 a = xi[i];
        a.x *= rt; a.y *= rt; a.z *= rt; a.w *= rt;
        xo[i] = a;
    }
}

// ----------------------------------------------------------------
extern "C" void kernel_launch(void* const* d_in, const int* in_sizes, int n_in,
                              void* d_out, int out_size, void* d_ws, size_t ws_size,
                              hipStream_t stream)
{
    const float* x         = (const float*)d_in[0];
    const int*   edges     = (const int*)d_in[1];
    const int*   dist      = (const int*)d_in[2];
    const float* W_temp    = (const float*)d_in[3];
    const float* conf_coef = (const float*)d_in[4];
    const float* bias      = (const float*)d_in[5];
    const float* train_a   = (const float*)d_in[6];
    const float* dist1_a   = (const float*)d_in[7];

    const int N = in_sizes[2];
    const int E = in_sizes[1] / 2;
    const int* src = edges;      // edge_index[0]
    const int* dst = edges + E;  // edge_index[1]

    char* ws = (char*)d_ws;
    size_t off = 0;
    auto take = [&](size_t bytes) -> char* {
        char* p = ws + off;
        off = (off + bytes + 255) & ~(size_t)255;
        return p;
    };
    _Float16* xs       = (_Float16*)take((size_t)N * 64 * 2);  // sorted rows, f16
    _Float16* Wpad     = (_Float16*)take(16 * 64 * 2);         // padded W, f16
    float*    a_cl     = (float*)take((size_t)N * 4);
    float*    rcp_a    = (float*)take((size_t)N * 4);
    float*    conf     = (float*)take((size_t)N * 4);
    float*    temp_sc  = (float*)take((size_t)N * 8 * 4);
    float*    e_buf    = (float*)take((size_t)E * 4);
    char*     zbase    = ws + off;                              // zeroed region start
    unsigned* emax_key = (unsigned*)take((size_t)N * 4);
    float*    denom    = (float*)take((size_t)N * 4);
    unsigned* deg      = (unsigned*)take((size_t)N * 4);
    float*    dconf    = (float*)take((size_t)N * 4);
    float*    sim      = (float*)take((size_t)N * 8 * 4);
    char*     zend     = ws + off;
    long zwords = (long)(zend - zbase) / 4;

    zero_ws<<<(unsigned)((zwords + 255) / 256), 256, 0, stream>>>((unsigned*)zbase, zwords);
    node_prep<<<(N + 127) / 128, 128, 0, stream>>>(x, dist, train_a, dist1_a,
                                                   xs, a_cl, rcp_a, conf, N);
    prep_w<<<4, 256, 0, stream>>>(W_temp, Wpad);
    int tiles = (N + 15) / 16;
    temp_wmma<<<(tiles + 7) / 8, 256, 0, stream>>>(xs, Wpad, a_cl, temp_sc, N);
    edge_pass1<<<(E + 255) / 256, 256, 0, stream>>>(x, src, dst, rcp_a,
                                                    e_buf, emax_key, deg, E);
    edge_pass2<<<(E + 255) / 256, 256, 0, stream>>>(dst, emax_key, e_buf, denom, E);
    edge_pass3<<<(E + 255) / 256, 256, 0, stream>>>(src, dst, e_buf, denom, temp_sc,
                                                    conf, sim, dconf, E);
    finalize_k<<<(N + 255) / 256, 256, 0, stream>>>(x, sim, dconf, deg,
                                                    conf_coef, bias, (float*)d_out, N);
}
